// LinearTemporalChannel_77867757077204
// MI455X (gfx1250) — matrix operand order repaired and checked by reading, not compiled, not run on hardware
//
#include <hip/hip_runtime.h>
#include <cstdint>
#include <cstddef>

// ---------------- problem constants ----------------
#define BB    4
#define SS    8192
#define DIMD  1024
#define HH    8
#define CC    16          // 2*H
#define EE    64          // DIM / C
#define NCH   128         // S / CHUNK
#define CHUNK 64
#define EPSF  1e-8f
#define TWOPI 6.283185307179586f

// ---------------- bf16 helpers ----------------
typedef __attribute__((ext_vector_type(16))) __bf16 v16bf;
typedef __attribute__((ext_vector_type(8)))  __bf16 v8bf;
typedef __attribute__((ext_vector_type(8)))  float  v8f;

union Frag16 { v16bf v; v8bf h[2]; };

static __device__ inline __bf16 f2bf(float f) {
    union { float f; unsigned u; } x; x.f = f;
    unsigned r = (x.u + 0x7FFFu + ((x.u >> 16) & 1u)) >> 16;
    unsigned short s = (unsigned short)r;
    union { unsigned short s; __bf16 b; } y; y.s = s;
    return y.b;
}
static __device__ inline float bf2f(__bf16 b) {
    union { unsigned short s; __bf16 b; } y; y.b = b;
    union { unsigned u; float f; } x; x.u = ((unsigned)y.s) << 16;
    return x.f;
}

// ============================================================
// Kernel 1: temporal feature tables  q0,q1,k0,k1,d per (b,c,t)
// ============================================================
__global__ __launch_bounds__(256)
void feat_kernel(const int* __restrict__ ts_i, const int* __restrict__ nts_i,
                 const unsigned char* __restrict__ mask,
                 const float* __restrict__ gamma_param,
                 float* __restrict__ Q0, float* __restrict__ Q1,
                 float* __restrict__ K0, float* __restrict__ K1,
                 float* __restrict__ Dd) {
    int idx = blockIdx.x * blockDim.x + threadIdx.x;
    if (idx >= BB * SS) return;
    int b = idx / SS, t = idx - b * SS;
    float ts  = (float)ts_i[idx];
    float nts = (float)nts_i[idx];
    bool  m   = mask[idx] != 0;
    float sd  = (t == 0) ? 0.0f : fmaxf(ts - (float)ts_i[idx - 1], 0.0f);
    float delta = fmaxf(nts - ts, 0.0f);

    const float INTERVALS[8] = {1.f, 10.f, 100.f, 1000.f, 10000.f,
                                100000.f, 1000000.f, 10000000.f};
#pragma unroll
    for (int h = 0; h < HH; ++h) {
        float inter = INTERVALS[h];
        float sf = TWOPI / inter;
        float thk = fmodf(ts, inter) * sf;
        float thq = fmodf(nts, inter) * sf;
        float ck = cosf(thk), sk = sinf(thk);
        float cq = cosf(thq), sq = sinf(thq);
        float gamma = 1.0f / (1.0f + expf(-gamma_param[h]));
        float lg = logf(fmaxf(gamma, 1e-6f));
        float qd = expf(lg * delta * sf);
        float dstep = expf(lg * sd * sf);
        float deff = m ? fmaxf(dstep, EPSF) : 1.0f;

        float q0a = m ? ( sq * qd) : 0.0f;   // channel h   : (sin_q, -cos_q)
        float q1a = m ? (-cq * qd) : 0.0f;
        float q0b = m ? ( cq * qd) : 0.0f;   // channel h+H : (cos_q,  sin_q)
        float q1b = m ? ( sq * qd) : 0.0f;
        float k0v = m ? ck : 0.0f;
        float k1v = m ? sk : 0.0f;

        size_t ba = ((size_t)(b * CC + h) * SS) + t;
        size_t bb = ((size_t)(b * CC + h + HH) * SS) + t;
        Q0[ba] = q0a; Q1[ba] = q1a; K0[ba] = k0v; K1[ba] = k1v; Dd[ba] = deff;
        Q0[bb] = q0b; Q1[bb] = q1b; K0[bb] = k0v; K1[bb] = k1v; Dd[bb] = deff;
    }
}

// ============================================================
// Kernel 2: values = inputs @ Wv^T via split-bf16 WMMA
//   block: 256 thr (8 waves), tile 128x128, K-step 32
//   double-buffered LDS (2 x 32 KB), one barrier per K-step
//   wave grid 2(M)x4(N); each wave: 4 M-subtiles x 2 N-subtiles
// ============================================================
#define KC 32
#define KSTEPS (DIMD / KC)
__global__ __launch_bounds__(256)
void gemm_kernel(const float* __restrict__ A, const float* __restrict__ W,
                 float* __restrict__ Out) {
    __shared__ __align__(16) __bf16 ldsAhi[2][128 * KC];
    __shared__ __align__(16) __bf16 ldsAlo[2][128 * KC];
    __shared__ __align__(16) __bf16 ldsBhi[2][128 * KC];
    __shared__ __align__(16) __bf16 ldsBlo[2][128 * KC];

    const int tid  = threadIdx.x;
    const int wave = tid >> 5;
    const int lane = tid & 31;
    const int l16  = lane & 15;
    const bool hiHalf = (lane >= 16);

    const int m0 = blockIdx.y * 128;
    const int n0 = blockIdx.x * 128;
    const int wm = (wave >> 2) * 64;   // 0 or 64
    const int wn = (wave & 3) * 32;    // 0,32,64,96

    const int lrow = tid >> 1;           // 0..127
    const int lseg = (tid & 1) * 16;     // 0 or 16
    const int lbase = lrow * KC + lseg;

    const float* apRow = A + (size_t)(m0 + lrow) * DIMD + lseg;
    const float* bpRow = W + (size_t)(n0 + lrow) * DIMD + lseg;

    float aReg[16], bReg[16];

    // ---- staging load of one K-tile into registers ----
    auto loadTile = [&](int kk) {
#pragma unroll
        for (int j = 0; j < 16; j += 4) {
            float4 av = *(const float4*)(apRow + kk + j);
            float4 bv = *(const float4*)(bpRow + kk + j);
            aReg[j + 0] = av.x; aReg[j + 1] = av.y;
            aReg[j + 2] = av.z; aReg[j + 3] = av.w;
            bReg[j + 0] = bv.x; bReg[j + 1] = bv.y;
            bReg[j + 2] = bv.z; bReg[j + 3] = bv.w;
        }
    };
    // ---- fp32 -> (hi,lo) bf16 split, store to LDS buffer ----
    auto storeTile = [&](int buf) {
#pragma unroll
        for (int j = 0; j < 16; ++j) {
            int li = lbase + j;
            __bf16 ah = f2bf(aReg[j]);
            ldsAhi[buf][li] = ah;
            ldsAlo[buf][li] = f2bf(aReg[j] - bf2f(ah));
            __bf16 bh = f2bf(bReg[j]);
            ldsBhi[buf][li] = bh;
            ldsBlo[buf][li] = f2bf(bReg[j] - bf2f(bh));
        }
    };

    v8f acc[4][2];
#pragma unroll
    for (int mi = 0; mi < 4; ++mi)
#pragma unroll
        for (int ni = 0; ni < 2; ++ni)
#pragma unroll
            for (int i = 0; i < 8; ++i) acc[mi][ni][i] = 0.0f;

    // prologue: tile 0 into buffer 0
    loadTile(0);
    storeTile(0);
    __syncthreads();

    for (int s = 0; s < KSTEPS; ++s) {
        const int cur = s & 1;

        // issue next tile's global loads early (hidden under WMMAs)
        if (s + 1 < KSTEPS) loadTile((s + 1) * KC);

        // ---- compute from buffer `cur` ----
        const __bf16* pAhi = ldsAhi[cur];
        const __bf16* pAlo = ldsAlo[cur];
        const __bf16* pBhi = ldsBhi[cur];
        const __bf16* pBlo = ldsBlo[cur];

        Frag16 bh[2], bl[2];
#pragma unroll
        for (int ni = 0; ni < 2; ++ni) {
            int n = wn + ni * 16 + l16;
            int baseB = n * KC + (hiHalf ? 16 : 0);
            bh[ni].h[0] = *(const v8bf*)&pBhi[baseB];
            bh[ni].h[1] = *(const v8bf*)&pBhi[baseB + 8];
            bl[ni].h[0] = *(const v8bf*)&pBlo[baseB];
            bl[ni].h[1] = *(const v8bf*)&pBlo[baseB + 8];
        }

#pragma unroll
        for (int mi = 0; mi < 4; ++mi) {
            int r = wm + mi * 16 + l16;
            int baseA = r * KC + (hiHalf ? 16 : 0);
            Frag16 ah, al;
            ah.h[0] = *(const v8bf*)&pAhi[baseA];
            ah.h[1] = *(const v8bf*)&pAhi[baseA + 8];
            al.h[0] = *(const v8bf*)&pAlo[baseA];
            al.h[1] = *(const v8bf*)&pAlo[baseA + 8];
#pragma unroll
            for (int ni = 0; ni < 2; ++ni) {
                acc[mi][ni] = __builtin_amdgcn_wmma_f32_16x16x32_bf16(
                    false, ah.v, false, bh[ni].v, (short)0, acc[mi][ni], false, false);
                acc[mi][ni] = __builtin_amdgcn_wmma_f32_16x16x32_bf16(
                    false, ah.v, false, bl[ni].v, (short)0, acc[mi][ni], false, false);
                acc[mi][ni] = __builtin_amdgcn_wmma_f32_16x16x32_bf16(
                    false, al.v, false, bh[ni].v, (short)0, acc[mi][ni], false, false);
            }
        }

        // stage next tile into the other buffer (read by next iteration)
        if (s + 1 < KSTEPS) storeTile(1 - cur);
        __syncthreads();
    }

    // store: VGPR i -> M = i (+8 for upper lanes), N = l16
#pragma unroll
    for (int mi = 0; mi < 4; ++mi) {
#pragma unroll
        for (int ni = 0; ni < 2; ++ni) {
            int col = n0 + wn + ni * 16 + l16;
#pragma unroll
            for (int i = 0; i < 8; ++i) {
                int row = m0 + wm + mi * 16 + i + (hiHalf ? 8 : 0);
                Out[(size_t)row * DIMD + col] = acc[mi][ni][i];
            }
        }
    }
}

// ============================================================
// Kernel 3: per-chunk aggregates  A[d=2][e], P  (64-step recurrence)
//   grid = B*C*NCH blocks, 64 threads (e lanes)
// ============================================================
__global__ __launch_bounds__(64)
void chunk_agg_kernel(const float* __restrict__ vals,
                      const float* __restrict__ K0, const float* __restrict__ K1,
                      const float* __restrict__ Dd,
                      float* __restrict__ A0, float* __restrict__ A1,
                      float* __restrict__ Pp) {
    int bid = blockIdx.x;
    int n = bid % NCH;
    int c = (bid / NCH) % CC;
    int b = bid / (NCH * CC);
    int e = threadIdx.x;

    size_t tb = (size_t)(b * CC + c) * SS + (size_t)n * CHUNK;
    float T0 = 0.0f, T1 = 0.0f, P = 1.0f;
#pragma unroll 4
    for (int l = 0; l < CHUNK; ++l) {
        int t = n * CHUNK + l;
        float d  = Dd[tb + l];
        float k0 = K0[tb + l];
        float k1 = K1[tb + l];
        float v  = vals[((size_t)(b * SS + t) * CC + c) * EE + e];
        T0 = fmaf(d, T0, k0 * v);
        T1 = fmaf(d, T1, k1 * v);
        P *= d;
    }
    size_t ob = ((size_t)(b * CC + c) * NCH + n) * EE + e;
    A0[ob] = T0;
    A1[ob] = T1;
    if (e == 0) Pp[(size_t)(b * CC + c) * NCH + n] = P;
}

// ============================================================
// Kernel 4: affine scan over chunks -> chunk-boundary states
//   grid = B*C blocks, 128 threads (d = tid>>6, e = tid&63)
// ============================================================
__global__ __launch_bounds__(128)
void scan_kernel(const float* __restrict__ A0, const float* __restrict__ A1,
                 const float* __restrict__ Pp, float* __restrict__ St) {
    int bc = blockIdx.x;             // b*C + c
    int d = threadIdx.x >> 6;
    int e = threadIdx.x & 63;
    const float* Ad = (d == 0) ? A0 : A1;
    float s = 0.0f;
    for (int n = 0; n < NCH; ++n) {
        St[(((size_t)bc * NCH + n) * 2 + d) * EE + e] = s;
        float P = Pp[(size_t)bc * NCH + n];
        float A = Ad[((size_t)bc * NCH + n) * EE + e];
        s = fmaf(P, s, A);
    }
}

// ============================================================
// Kernel 5: output — recurrence from boundary state, emits
//   out = attn*alpha[c] + values*beta[c]
//   grid = B*C*NCH blocks, 64 threads (e lanes)
// ============================================================
__global__ __launch_bounds__(64)
void out_kernel(const float* __restrict__ vals,
                const float* __restrict__ Q0, const float* __restrict__ Q1,
                const float* __restrict__ K0, const float* __restrict__ K1,
                const float* __restrict__ Dd, const float* __restrict__ St,
                const float* __restrict__ alpha, const float* __restrict__ beta,
                float* __restrict__ out) {
    int bid = blockIdx.x;
    int n = bid % NCH;
    int c = (bid / NCH) % CC;
    int b = bid / (NCH * CC);
    int e = threadIdx.x;

    size_t bc = (size_t)(b * CC + c);
    float s0 = St[((bc * NCH + n) * 2 + 0) * EE + e];
    float s1 = St[((bc * NCH + n) * 2 + 1) * EE + e];
    float al = alpha[c], be = beta[c];
    size_t tb = bc * SS + (size_t)n * CHUNK;

#pragma unroll 4
    for (int l = 0; l < CHUNK; ++l) {
        int t = n * CHUNK + l;
        float d  = Dd[tb + l];
        float k0 = K0[tb + l];
        float k1 = K1[tb + l];
        float q0 = Q0[tb + l];
        float q1 = Q1[tb + l];
        size_t vi = ((size_t)(b * SS + t) * CC + c) * EE + e;
        float v = vals[vi];
        s0 = fmaf(d, s0, k0 * v);
        s1 = fmaf(d, s1, k1 * v);
        float attn = fmaf(q0, s0, q1 * s1);
        out[vi] = fmaf(attn, al, v * be);
    }
}

// ============================================================
extern "C" void kernel_launch(void* const* d_in, const int* in_sizes, int n_in,
                              void* d_out, int out_size, void* d_ws, size_t ws_size,
                              hipStream_t stream) {
    const float*         inputs = (const float*)d_in[0];
    const float*         Wv     = (const float*)d_in[1];
    const float*         gp     = (const float*)d_in[2];
    const float*         alpha  = (const float*)d_in[3];
    const float*         beta   = (const float*)d_in[4];
    const int*           ts     = (const int*)d_in[5];
    const int*           nts    = (const int*)d_in[6];
    const unsigned char* mask   = (const unsigned char*)d_in[7];

    float* ws = (float*)d_ws;
    const size_t N_VALS = (size_t)BB * SS * DIMD;      // 33.5M floats
    const size_t N_TAB  = (size_t)BB * CC * SS;        // 524288 floats each
    const size_t N_AGG  = (size_t)BB * CC * NCH * EE;  // 524288
    const size_t N_P    = (size_t)BB * CC * NCH;       // 8192
    const size_t N_ST   = (size_t)BB * CC * NCH * 2 * EE;

    float* vals = ws;                   size_t off = N_VALS;
    float* Q0 = ws + off; off += N_TAB;
    float* Q1 = ws + off; off += N_TAB;
    float* K0 = ws + off; off += N_TAB;
    float* K1 = ws + off; off += N_TAB;
    float* Dd = ws + off; off += N_TAB;
    float* A0 = ws + off; off += N_AGG;
    float* A1 = ws + off; off += N_AGG;
    float* Pp = ws + off; off += N_P;
    float* St = ws + off; off += N_ST;
    (void)ws_size; (void)in_sizes; (void)n_in; (void)out_size;

    // 1. feature tables
    feat_kernel<<<(BB * SS + 255) / 256, 256, 0, stream>>>(
        ts, nts, mask, gp, Q0, Q1, K0, K1, Dd);

    // 2. WMMA GEMM: values = inputs @ Wv^T
    gemm_kernel<<<dim3(DIMD / 128, (BB * SS) / 128), 256, 0, stream>>>(
        inputs, Wv, vals);

    // 3. chunk aggregates
    chunk_agg_kernel<<<BB * CC * NCH, 64, 0, stream>>>(
        vals, K0, K1, Dd, A0, A1, Pp);

    // 4. inter-chunk state scan
    scan_kernel<<<BB * CC, 128, 0, stream>>>(A0, A1, Pp, St);

    // 5. outputs
    out_kernel<<<BB * CC * NCH, 64, 0, stream>>>(
        vals, Q0, Q1, K0, K1, Dd, St, alpha, beta, (float*)d_out);
}
